// EdgeRankingGNN2_41875931136402
// MI455X (gfx1250) — matrix operand order repaired and checked by reading, not compiled, not run on hardware
//
#include <hip/hip_runtime.h>
#include <hip/hip_bf16.h>
#include <math.h>

#define HD 128           // hidden size H
#define NB 16            // number of graphs B

typedef __attribute__((ext_vector_type(16))) _Float16 v16h;
typedef __attribute__((ext_vector_type(8)))  float    v8f;

// ---------------------------------------------------------------------------
// WMMA fragment helpers (CDNA5 wave32 layouts, CDNA5 ISA §7.12.2)
// ---------------------------------------------------------------------------

// A fragment: 16x32 f16 tile from row-major LDS, stride lda (halves).
// lanes 0-15: row M=lane, K halves {0..7,16..23}; lanes 16-31: +8 on K.
// Each half-wave segment is 8 contiguous halves -> ds_load_b128.
__device__ __forceinline__ v16h load_a_frag(const _Float16* A, int lda, int kbase, int lane) {
  const int m    = lane & 15;
  const int koff = (lane >> 4) << 3;           // 0 or 8
  v16h a;
#pragma unroll
  for (int j = 0; j < 8; ++j) {
    const int k = kbase + ((j & 4) << 2) + koff + ((j & 3) << 1);
    a[2 * j]     = A[m * lda + k];
    a[2 * j + 1] = A[m * lda + k + 1];
  }
  return a;
}

// B fragment from PRE-SWIZZLED weights: each (32x16 chunk, lane) owns 16
// contiguous halves (32 bytes) -> two global_load_b128, zero packing ops.
__device__ __forceinline__ v16h load_b_sw(const _Float16* __restrict__ Bsw,
                                          int NT, int kk, int nt, int lane) {
  const _Float16* p = Bsw + (((size_t)(kk * NT + nt) * 32 + lane) << 4);
  return *(const v16h*)p;
}

__device__ __forceinline__ v8f wmma16(v16h a, v16h b, v8f c) {
  return __builtin_amdgcn_wmma_f32_16x16x32_f16(false, a, false, b, (short)0, c, false, false);
}

// ---------------------------------------------------------------------------
// Weight conversion: f32 row-major W[K][N] -> f16 WMMA-B fragment order.
// out[((kk*NT+nt)*32+lane)*16 + 2j+{0,1}] = W[kk*32+(j&4)*4+(lane>>4)*8+(j&3)*2+{0,1}][nt*16+(lane&15)]
// ---------------------------------------------------------------------------
__global__ void cvt_swizzle_b(const float* __restrict__ in, _Float16* __restrict__ out,
                              int K, int Nn) {
  int i = blockIdx.x * blockDim.x + threadIdx.x;
  int total = (K >> 5) * (Nn >> 4) * 256;
  if (i >= total) return;
  int j = i & 7;
  int lane = (i >> 3) & 31;
  int chunk = i >> 8;
  int NT = Nn >> 4;
  int kk = chunk / NT, nt = chunk % NT;
  int n = nt * 16 + (lane & 15);
  int k = kk * 32 + ((j & 4) << 2) + ((lane >> 4) << 3) + ((j & 3) << 1);
  out[(size_t)i * 2]     = (_Float16)in[(size_t)k * Nn + n];
  out[(size_t)i * 2 + 1] = (_Float16)in[(size_t)(k + 1) * Nn + n];
}

__global__ void cvt_f32_f16(const float* __restrict__ in, _Float16* __restrict__ out, int n) {
  int i = blockIdx.x * blockDim.x + threadIdx.x;
  if (i < n) out[i] = (_Float16)in[i];
}

// ---------------------------------------------------------------------------
// Encoder: y = LN(relu(x@W1+b1)@W2+b2). x is [rows,3]. 16-row tile per block.
// ---------------------------------------------------------------------------
__global__ __launch_bounds__(256)
void encoder_mlp_ln(const float* __restrict__ xin,
                    const float* __restrict__ w1, const float* __restrict__ b1,
                    const _Float16* __restrict__ w2s, const float* __restrict__ b2,
                    const float* __restrict__ gam, const float* __restrict__ bet,
                    int rows, float* __restrict__ out_f, _Float16* __restrict__ out_h) {
  __shared__ float    xi[16][4];
  __shared__ _Float16 A[16][136];
  __shared__ float    Z[16][136];
  const int tid  = threadIdx.x;
  const int base = blockIdx.x * 16;

  if (tid < 48) {
    int r = tid / 3, k = tid % 3;
    int gr = min(base + r, rows - 1);
    xi[r][k] = xin[(size_t)gr * 3 + k];
  }
  __syncthreads();

  // layer 1: [16,3]@[3,128] + relu -> A (f16), 8 consecutive cols per thread
  {
    const int r = tid >> 4, c0 = (tid & 15) * 8;
    _Float16 tmp[8];
#pragma unroll
    for (int i = 0; i < 8; ++i) {
      int c = c0 + i;
      float v = b1[c] + xi[r][0] * w1[c] + xi[r][1] * w1[HD + c] + xi[r][2] * w1[2 * HD + c];
      tmp[i] = (_Float16)fmaxf(v, 0.0f);
    }
    *(uint4*)&A[r][c0] = *(const uint4*)tmp;
  }
  __syncthreads();

  // layer 2: [16,128]@[128,128] via WMMA
  const int lane = tid & 31, wave = tid >> 5;
  v8f acc = {};
#pragma unroll
  for (int kk = 0; kk < 4; ++kk)
    acc = wmma16(load_a_frag(&A[0][0], 136, 32 * kk, lane),
                 load_b_sw(w2s, 8, kk, wave, lane), acc);
  {
    const int col = wave * 16 + (lane & 15);
    const int rb  = (lane >> 4) * 8;
    const float bc = b2[col];
#pragma unroll
    for (int j = 0; j < 8; ++j) Z[rb + j][col] = acc[j] + bc;
  }
  __syncthreads();

  // LayerNorm (16 threads per row, shfl-16 reduction)
  {
    const int r = tid >> 4, sub = tid & 15;
    float v[8], s = 0.f, sq = 0.f;
#pragma unroll
    for (int i = 0; i < 8; ++i) { v[i] = Z[r][sub * 8 + i]; s += v[i]; sq += v[i] * v[i]; }
#pragma unroll
    for (int o = 8; o >= 1; o >>= 1) { s += __shfl_xor(s, o, 16); sq += __shfl_xor(sq, o, 16); }
    const float mu = s * (1.0f / HD);
    const float rstd = rsqrtf(sq * (1.0f / HD) - mu * mu + 1e-5f);
    const int gr = base + r;
    if (gr < rows) {
#pragma unroll
      for (int i = 0; i < 8; ++i) {
        int c = sub * 8 + i;
        float y = (v[i] - mu) * rstd * gam[c] + bet[c];
        if (out_f) out_f[(size_t)gr * HD + c] = y;
        if (out_h) out_h[(size_t)gr * HD + c] = (_Float16)y;
      }
    }
  }
}

// ---------------------------------------------------------------------------
// GINE message + scatter: agg[dst] += relu(h[src] + e)   (memory bound)
// ---------------------------------------------------------------------------
__global__ void gine_msg(const float* __restrict__ hf, const _Float16* __restrict__ eh,
                         const int* __restrict__ src, const int* __restrict__ dst,
                         float* __restrict__ agg, int E) {
  int i = blockIdx.x * blockDim.x + threadIdx.x;
  if (i >= E * 32) return;
  int e = i >> 5, c0 = (i & 31) * 4;
  int s = src[e], d = dst[e];
  const float4 hv = *(const float4*)&hf[(size_t)s * HD + c0];
  const _Float16* ep = &eh[(size_t)e * HD + c0];
  float m0 = fmaxf(hv.x + (float)ep[0], 0.0f);
  float m1 = fmaxf(hv.y + (float)ep[1], 0.0f);
  float m2 = fmaxf(hv.z + (float)ep[2], 0.0f);
  float m3 = fmaxf(hv.w + (float)ep[3], 0.0f);
  float* ap = &agg[(size_t)d * HD + c0];
  atomicAdd(ap + 0, m0);
  atomicAdd(ap + 1, m1);
  atomicAdd(ap + 2, m2);
  atomicAdd(ap + 3, m3);
}

// ---------------------------------------------------------------------------
// GINE MLP: h = LN(relu(z@W1+b1)@W2+b2), z=(1+eps)h+agg. Optional trailing relu.
// ---------------------------------------------------------------------------
__global__ __launch_bounds__(256)
void gine_mlp(const float* __restrict__ agg, const float* __restrict__ eps_p,
              const _Float16* __restrict__ w1s, const float* __restrict__ b1,
              const _Float16* __restrict__ w2s, const float* __restrict__ b2,
              const float* __restrict__ gam, const float* __restrict__ bet,
              int rows, int relu_out, float* __restrict__ hf, _Float16* __restrict__ hh) {
  __shared__ _Float16 A[16][136];
  __shared__ float    Z[16][136];
  const int tid = threadIdx.x, base = blockIdx.x * 16;
  const float ep1 = 1.0f + eps_p[0];

  {
    const int r = tid >> 4, c0 = (tid & 15) * 8;
    const int gr = min(base + r, rows - 1);
    _Float16 tmp[8];
#pragma unroll
    for (int i = 0; i < 8; ++i) {
      int c = c0 + i;
      tmp[i] = (_Float16)(ep1 * hf[(size_t)gr * HD + c] + agg[(size_t)gr * HD + c]);
    }
    *(uint4*)&A[r][c0] = *(const uint4*)tmp;
  }
  __syncthreads();

  const int lane = tid & 31, wave = tid >> 5;
  const int col = wave * 16 + (lane & 15), rb = (lane >> 4) * 8;

  v8f c1 = {};
#pragma unroll
  for (int kk = 0; kk < 4; ++kk)
    c1 = wmma16(load_a_frag(&A[0][0], 136, 32 * kk, lane),
                load_b_sw(w1s, 8, kk, wave, lane), c1);
  __syncthreads();                       // all reads of A done
  {
    const float bc = b1[col];
#pragma unroll
    for (int j = 0; j < 8; ++j) A[rb + j][col] = (_Float16)fmaxf(c1[j] + bc, 0.0f);
  }
  __syncthreads();

  v8f c2 = {};
#pragma unroll
  for (int kk = 0; kk < 4; ++kk)
    c2 = wmma16(load_a_frag(&A[0][0], 136, 32 * kk, lane),
                load_b_sw(w2s, 8, kk, wave, lane), c2);
  {
    const float bc = b2[col];
#pragma unroll
    for (int j = 0; j < 8; ++j) Z[rb + j][col] = c2[j] + bc;
  }
  __syncthreads();

  {
    const int r = tid >> 4, sub = tid & 15;
    float v[8], s = 0.f, sq = 0.f;
#pragma unroll
    for (int i = 0; i < 8; ++i) { v[i] = Z[r][sub * 8 + i]; s += v[i]; sq += v[i] * v[i]; }
#pragma unroll
    for (int o = 8; o >= 1; o >>= 1) { s += __shfl_xor(s, o, 16); sq += __shfl_xor(sq, o, 16); }
    const float mu = s * (1.0f / HD);
    const float rstd = rsqrtf(sq * (1.0f / HD) - mu * mu + 1e-5f);
    const int gr = base + r;
    if (gr < rows) {
#pragma unroll
      for (int i = 0; i < 8; ++i) {
        int c = sub * 8 + i;
        float y = (v[i] - mu) * rstd * gam[c] + bet[c];
        if (relu_out) y = fmaxf(y, 0.0f);
        hf[(size_t)gr * HD + c] = y;
        hh[(size_t)gr * HD + c] = (_Float16)y;
      }
    }
  }
}

// ---------------------------------------------------------------------------
// Global mean pool: atomics into gsum/gcnt
// ---------------------------------------------------------------------------
__global__ void pool_sum(const float* __restrict__ hf, const int* __restrict__ batch,
                         float* __restrict__ gsum, float* __restrict__ gcnt, int N) {
  int i = blockIdx.x * blockDim.x + threadIdx.x;
  if (i >= N * 32) return;
  int n = i >> 5, c0 = (i & 31) * 4;
  int b = batch[n];
  const float4 hv = *(const float4*)&hf[(size_t)n * HD + c0];
  float* gp = &gsum[(size_t)b * HD + c0];
  atomicAdd(gp + 0, hv.x);
  atomicAdd(gp + 1, hv.y);
  atomicAdd(gp + 2, hv.z);
  atomicAdd(gp + 3, hv.w);
  if ((i & 31) == 0) atomicAdd(&gcnt[b], 1.0f);
}

// g = LN(relu((gsum/cnt)@gp_w + gp_b)); one block, B=16 rows == one WMMA tile.
__global__ __launch_bounds__(256)
void pool_mlp(const float* __restrict__ gsum, const float* __restrict__ gcnt,
              const _Float16* __restrict__ ws, const float* __restrict__ b,
              const float* __restrict__ gam, const float* __restrict__ bet,
              _Float16* __restrict__ gh) {
  __shared__ _Float16 A[16][136];
  __shared__ float    Z[16][136];
  const int tid = threadIdx.x;
  {
    const int r = tid >> 4, c0 = (tid & 15) * 8;
    _Float16 tmp[8];
#pragma unroll
    for (int i = 0; i < 8; ++i)
      tmp[i] = (_Float16)(gsum[(size_t)r * HD + c0 + i] / fmaxf(gcnt[r], 1.0f));
    *(uint4*)&A[r][c0] = *(const uint4*)tmp;
  }
  __syncthreads();
  const int lane = tid & 31, wave = tid >> 5;
  const int col = wave * 16 + (lane & 15), rb = (lane >> 4) * 8;
  v8f acc = {};
#pragma unroll
  for (int kk = 0; kk < 4; ++kk)
    acc = wmma16(load_a_frag(&A[0][0], 136, 32 * kk, lane),
                 load_b_sw(ws, 8, kk, wave, lane), acc);
  {
    const float bc = b[col];
#pragma unroll
    for (int j = 0; j < 8; ++j) Z[rb + j][col] = fmaxf(acc[j] + bc, 0.0f);
  }
  __syncthreads();
  {
    const int r = tid >> 4, sub = tid & 15;
    float v[8], s = 0.f, sq = 0.f;
#pragma unroll
    for (int i = 0; i < 8; ++i) { v[i] = Z[r][sub * 8 + i]; s += v[i]; sq += v[i] * v[i]; }
#pragma unroll
    for (int o = 8; o >= 1; o >>= 1) { s += __shfl_xor(s, o, 16); sq += __shfl_xor(sq, o, 16); }
    const float mu = s * (1.0f / HD);
    const float rstd = rsqrtf(sq * (1.0f / HD) - mu * mu + 1e-5f);
#pragma unroll
    for (int i = 0; i < 8; ++i) {
      int c = sub * 8 + i;
      gh[(size_t)r * HD + c] = (_Float16)((v[i] - mu) * rstd * gam[c] + bet[c]);
    }
  }
}

// ---------------------------------------------------------------------------
// Fused edge predictor, 64 edges per block (4 row-tiles) for 4x weight reuse:
// comb=[h[src],h[dst],g[batch[src]],e] (64x512 LDS), tanh@W1 -> tanh@W2 -> sigmoid@w3.
// S2 overlays dead C to keep LDS ~100KB (3 blocks/WGP of 320KB).
// ---------------------------------------------------------------------------
__global__ __launch_bounds__(256)
void edge_pred(const _Float16* __restrict__ hh, const _Float16* __restrict__ eh,
               const _Float16* __restrict__ gh,
               const int* __restrict__ src, const int* __restrict__ dst,
               const int* __restrict__ batch,
               const _Float16* __restrict__ w1s, const float* __restrict__ b1,
               const _Float16* __restrict__ w2s, const float* __restrict__ b2,
               const _Float16* __restrict__ w3h, const float* __restrict__ b3,
               int E, float* __restrict__ out) {
  __shared__ _Float16 C[64][520];    // 66.6 KB (dead after matmul1)
  __shared__ _Float16 S1[64][264];   // 33.8 KB
  _Float16 (*S2)[136] = (_Float16 (*)[136])&C[0][0];   // 17.4 KB overlay
  const int tid = threadIdx.x, base = blockIdx.x * 64;
  const int lane = tid & 31, wave = tid >> 5;

  // gather comb tile: (row, 32-col segment) -> 4x uint4; h/g tables L2-resident
  {
    const int r0 = tid >> 4, seg = tid & 15, cbase = seg * 32;
#pragma unroll
    for (int rt = 0; rt < 4; ++rt) {
      const int r = rt * 16 + r0;
      const int e = min(base + r, E - 1);
      const int s = src[e], d = dst[e], bg = batch[s];
      const _Float16* p;
      if (cbase < 128)      p = hh + (size_t)s  * HD + cbase;
      else if (cbase < 256) p = hh + (size_t)d  * HD + (cbase - 128);
      else if (cbase < 384) p = gh + (size_t)bg * HD + (cbase - 256);
      else                  p = eh + (size_t)e  * HD + (cbase - 384);
      const uint4* ps = (const uint4*)p;
      uint4* pd = (uint4*)&C[r][cbase];
      pd[0] = ps[0]; pd[1] = ps[1]; pd[2] = ps[2]; pd[3] = ps[3];
    }
  }
  __syncthreads();

  // s1 = tanh(comb @ W1 + b1): [64,512]@[512,256]; wave -> 2 col-tiles,
  // B fragment loaded once and reused across 4 row-tiles (4 indep WMMA chains)
  for (int t = wave; t < 16; t += 8) {
    v8f a0 = {}, a1 = {}, a2 = {}, a3 = {};
    for (int kk = 0; kk < 16; ++kk) {
      const v16h b = load_b_sw(w1s, 16, kk, t, lane);
      a0 = wmma16(load_a_frag(&C[0][0],  520, 32 * kk, lane), b, a0);
      a1 = wmma16(load_a_frag(&C[16][0], 520, 32 * kk, lane), b, a1);
      a2 = wmma16(load_a_frag(&C[32][0], 520, 32 * kk, lane), b, a2);
      a3 = wmma16(load_a_frag(&C[48][0], 520, 32 * kk, lane), b, a3);
    }
    const int col = 16 * t + (lane & 15), rb = (lane >> 4) * 8;
    const float bc = b1[col];
#pragma unroll
    for (int j = 0; j < 8; ++j) {
      S1[rb + j][col]      = (_Float16)tanhf(a0[j] + bc);
      S1[16 + rb + j][col] = (_Float16)tanhf(a1[j] + bc);
      S1[32 + rb + j][col] = (_Float16)tanhf(a2[j] + bc);
      S1[48 + rb + j][col] = (_Float16)tanhf(a3[j] + bc);
    }
  }
  __syncthreads();   // C now dead; S2 overlay becomes live

  // s2 = tanh(s1 @ W2 + b2): [64,256]@[256,128]; wave -> 1 col-tile, 4 row-tiles
  {
    v8f a0 = {}, a1 = {}, a2 = {}, a3 = {};
    for (int kk = 0; kk < 8; ++kk) {
      const v16h b = load_b_sw(w2s, 8, kk, wave, lane);
      a0 = wmma16(load_a_frag(&S1[0][0],  264, 32 * kk, lane), b, a0);
      a1 = wmma16(load_a_frag(&S1[16][0], 264, 32 * kk, lane), b, a1);
      a2 = wmma16(load_a_frag(&S1[32][0], 264, 32 * kk, lane), b, a2);
      a3 = wmma16(load_a_frag(&S1[48][0], 264, 32 * kk, lane), b, a3);
    }
    const int col = 16 * wave + (lane & 15), rb = (lane >> 4) * 8;
    const float bc = b2[col];
#pragma unroll
    for (int j = 0; j < 8; ++j) {
      S2[rb + j][col]      = (_Float16)tanhf(a0[j] + bc);
      S2[16 + rb + j][col] = (_Float16)tanhf(a1[j] + bc);
      S2[32 + rb + j][col] = (_Float16)tanhf(a2[j] + bc);
      S2[48 + rb + j][col] = (_Float16)tanhf(a3[j] + bc);
    }
  }
  __syncthreads();

  // out = sigmoid(s2 @ w3 + b3): per-row dot, shfl-16 reduce
  {
    const int r0 = tid >> 4, sub = tid & 15;
#pragma unroll
    for (int rt = 0; rt < 4; ++rt) {
      const int r = rt * 16 + r0;
      float s = 0.f;
#pragma unroll
      for (int i = 0; i < 8; ++i) {
        int c = sub * 8 + i;
        s += (float)S2[r][c] * (float)w3h[c];
      }
#pragma unroll
      for (int o = 8; o >= 1; o >>= 1) s += __shfl_xor(s, o, 16);
      const int e = base + r;
      if (sub == 0 && e < E) out[e] = 1.0f / (1.0f + expf(-(s + b3[0])));
    }
  }
}

// ---------------------------------------------------------------------------
// Host launcher
// ---------------------------------------------------------------------------
extern "C" void kernel_launch(void* const* d_in, const int* in_sizes, int n_in,
                              void* d_out, int out_size, void* d_ws, size_t ws_size,
                              hipStream_t stream) {
  const float* x         = (const float*)d_in[0];
  const float* edge_attr = (const float*)d_in[1];
  const int*   edge_idx  = (const int*)d_in[2];
  const int*   batch     = (const int*)d_in[3];
  const float* ne_w1 = (const float*)d_in[4];
  const float* ne_b1 = (const float*)d_in[5];
  const float* ne_w2 = (const float*)d_in[6];
  const float* ne_b2 = (const float*)d_in[7];
  const float* ne_g  = (const float*)d_in[8];
  const float* ne_be = (const float*)d_in[9];
  const float* ee_w1 = (const float*)d_in[10];
  const float* ee_b1 = (const float*)d_in[11];
  const float* ee_w2 = (const float*)d_in[12];
  const float* ee_b2 = (const float*)d_in[13];
  const float* ee_g  = (const float*)d_in[14];
  const float* ee_be = (const float*)d_in[15];
  const float* gin_eps = (const float*)d_in[16];
  const float* gin_w1  = (const float*)d_in[17];
  const float* gin_b1  = (const float*)d_in[18];
  const float* gin_w2  = (const float*)d_in[19];
  const float* gin_b2  = (const float*)d_in[20];
  const float* gin_g   = (const float*)d_in[21];
  const float* gin_be  = (const float*)d_in[22];
  const float* gp_w  = (const float*)d_in[23];
  const float* gp_b  = (const float*)d_in[24];
  const float* gp_g  = (const float*)d_in[25];
  const float* gp_be = (const float*)d_in[26];
  const float* ep_w1 = (const float*)d_in[27];
  const float* ep_b1 = (const float*)d_in[28];
  const float* ep_w2 = (const float*)d_in[29];
  const float* ep_b2 = (const float*)d_in[30];
  const float* ep_w3 = (const float*)d_in[31];
  const float* ep_b3 = (const float*)d_in[32];

  const int N = in_sizes[3];
  const int E = in_sizes[1] / 3;
  const int* src = edge_idx;       // edge_index[0]
  const int* dst = edge_idx + E;   // edge_index[1]

  // workspace carve-out
  char* ws = (char*)d_ws;
  size_t off = 0;
  auto take = [&](size_t bytes) -> char* {
    char* p = ws + off;
    off += (bytes + 255) & ~(size_t)255;
    return p;
  };
  _Float16* eh   = (_Float16*)take((size_t)E * HD * 2);
  float*    hf   = (float*)take((size_t)N * HD * 4);
  _Float16* hh   = (_Float16*)take((size_t)N * HD * 2);
  float*    agg  = (float*)take((size_t)N * HD * 4);
  float*    gsum = (float*)take((size_t)NB * HD * 4);
  float*    gcnt = (float*)take((size_t)NB * 4);
  _Float16* gh   = (_Float16*)take((size_t)NB * HD * 2);
  _Float16* ne_w2s  = (_Float16*)take((size_t)HD * HD * 2);
  _Float16* ee_w2s  = (_Float16*)take((size_t)HD * HD * 2);
  _Float16* gin_w1s = (_Float16*)take((size_t)2 * HD * HD * 2);
  _Float16* gin_w2s = (_Float16*)take((size_t)2 * HD * HD * 2);
  _Float16* gp_ws   = (_Float16*)take((size_t)HD * HD * 2);
  _Float16* ep_w1s  = (_Float16*)take((size_t)4 * HD * 2 * HD * 2);
  _Float16* ep_w2s  = (_Float16*)take((size_t)2 * HD * HD * 2);
  _Float16* ep_w3h  = (_Float16*)take((size_t)HD * 2);

  auto cvtB = [&](const float* in, _Float16* outp, int K, int Nn) {
    int total = (K / 32) * (Nn / 16) * 256;
    cvt_swizzle_b<<<(total + 255) / 256, 256, 0, stream>>>(in, outp, K, Nn);
  };
  cvtB(ne_w2, ne_w2s, HD, HD);
  cvtB(ee_w2, ee_w2s, HD, HD);
  for (int l = 0; l < 2; ++l) {
    cvtB(gin_w1 + (size_t)l * HD * HD, gin_w1s + (size_t)l * HD * HD, HD, HD);
    cvtB(gin_w2 + (size_t)l * HD * HD, gin_w2s + (size_t)l * HD * HD, HD, HD);
  }
  cvtB(gp_w, gp_ws, HD, HD);
  cvtB(ep_w1, ep_w1s, 4 * HD, 2 * HD);
  cvtB(ep_w2, ep_w2s, 2 * HD, HD);
  cvt_f32_f16<<<1, 256, 0, stream>>>(ep_w3, ep_w3h, HD);

  // encoders
  encoder_mlp_ln<<<(N + 15) / 16, 256, 0, stream>>>(
      x, ne_w1, ne_b1, ne_w2s, ne_b2, ne_g, ne_be, N, hf, hh);
  encoder_mlp_ln<<<(E + 15) / 16, 256, 0, stream>>>(
      edge_attr, ee_w1, ee_b1, ee_w2s, ee_b2, ee_g, ee_be, E, (float*)nullptr, eh);

  // 2 GINE layers
  for (int l = 0; l < 2; ++l) {
    hipMemsetAsync(agg, 0, (size_t)N * HD * 4, stream);
    gine_msg<<<(E * 32 + 255) / 256, 256, 0, stream>>>(hf, eh, src, dst, agg, E);
    gine_mlp<<<(N + 15) / 16, 256, 0, stream>>>(
        agg, gin_eps + l,
        gin_w1s + (size_t)l * HD * HD, gin_b1 + (size_t)l * HD,
        gin_w2s + (size_t)l * HD * HD, gin_b2 + (size_t)l * HD,
        gin_g + (size_t)l * HD, gin_be + (size_t)l * HD,
        N, (l < 1) ? 1 : 0, hf, hh);
  }

  // global mean pool + processor
  hipMemsetAsync(gsum, 0, (size_t)NB * HD * 4, stream);
  hipMemsetAsync(gcnt, 0, (size_t)NB * 4, stream);
  pool_sum<<<(N * 32 + 255) / 256, 256, 0, stream>>>(hf, batch, gsum, gcnt, N);
  pool_mlp<<<1, 256, 0, stream>>>(gsum, gcnt, gp_ws, gp_b, gp_g, gp_be, gh);

  // fused edge predictor (64 edges / block)
  edge_pred<<<(E + 63) / 64, 256, 0, stream>>>(
      hh, eh, gh, src, dst, batch,
      ep_w1s, ep_b1, ep_w2s, ep_b2, ep_w3h, ep_b3, E, (float*)d_out);
}
